// AFM_54022098649750
// MI455X (gfx1250) — matrix operand order repaired and checked by reading, not compiled, not run on hardware
//
#include <hip/hip_runtime.h>
#include <hip/hip_bf16.h>

// ---------------------------------------------------------------------------
// Problem constants (from reference): B=2048, DIM=512, C=64, INTER=128
// ---------------------------------------------------------------------------
#define DIMX   512
#define CBR    64
#define INTERX 128
#define EPSX   1e-5f

#define BT      128   // b-tile rows per workgroup
#define KC      32    // k-chunk (matches bf16 WMMA K)
#define NTHREADS 256  // 8 wave32
#define SAPITCH 40    // bf16 pitch for 32-wide staged tiles (80B, 16B aligned)
#define HPITCH  136   // bf16 pitch for 128-wide tiles (272B, 16B aligned)

typedef __bf16 bf16;
typedef __attribute__((ext_vector_type(16))) __bf16 v16bf;
typedef __attribute__((ext_vector_type(8)))  __bf16 v8bf;
typedef __attribute__((ext_vector_type(8)))  float  v8f;

// A-matrix fragment (16x32 bf16, tile stored row-major M x K, pitch in elems).
// ISA layout: lanes 0-15 hold row M=lane, K in {0..7, 16..23}; lanes 16-31 hold
// K in {8..15, 24..31}.  -> two 16B LDS loads per lane.
__device__ __forceinline__ v16bf frag_a(const bf16* tile, int pitch, int lane) {
    const bf16* p = tile + (lane & 15) * pitch + ((lane >> 4) << 3);
    v8bf lo = *(const v8bf*)(p);
    v8bf hi = *(const v8bf*)(p + 16);
    return __builtin_shufflevector(lo, hi, 0,1,2,3,4,5,6,7,8,9,10,11,12,13,14,15);
}

// B-matrix fragment (32x16 bf16) loaded from B^T tile stored row-major N x K.
// ISA layout: lane holds column N=lane%16; lanes 0-15 K in {0..7, 16..23},
// lanes 16-31 K in {8..15, 24..31}.  -> two 16B LDS loads per lane.
__device__ __forceinline__ v16bf frag_b(const bf16* tile, int pitch, int lane) {
    const bf16* p = tile + (lane & 15) * pitch + ((lane >> 4) << 3);
    v8bf lo = *(const v8bf*)(p);
    v8bf hi = *(const v8bf*)(p + 16);
    return __builtin_shufflevector(lo, hi, 0,1,2,3,4,5,6,7,8,9,10,11,12,13,14,15);
}

__global__ __launch_bounds__(NTHREADS)
void AFM_54022098649750_kernel(const float* __restrict__ x,
                               const float* __restrict__ W1, const float* __restrict__ b1,
                               const float* __restrict__ g1, const float* __restrict__ be1,
                               const float* __restrict__ m1, const float* __restrict__ v1,
                               const float* __restrict__ W2, const float* __restrict__ b2,
                               const float* __restrict__ g2, const float* __restrict__ be2,
                               const float* __restrict__ m2, const float* __restrict__ v2,
                               float* __restrict__ out)
{
    const int c    = blockIdx.x;          // branch 0..63 (fast-varying -> L2 reuse of x)
    const int b0   = blockIdx.y * BT;     // b-tile base
    const int tid  = threadIdx.x;
    const int lane = tid & 31;
    const int wave = tid >> 5;            // 0..7, owns rows [16w, 16w+16)

    // ---------------- LDS ----------------
    __shared__ __attribute__((aligned(16))) float sScale1[INTERX];
    __shared__ __attribute__((aligned(16))) float sShift1[INTERX];
    __shared__ __attribute__((aligned(16))) float sScale2[DIMX];
    __shared__ __attribute__((aligned(16))) float sShift2[DIMX];
    __shared__ __attribute__((aligned(16))) bf16  sH[BT][HPITCH];              // 34816 B
    __shared__ __attribute__((aligned(16))) bf16  sStage[2 * BT * SAPITCH];    // 20480 B (union)
    bf16* sA = sStage;                 // [BT][SAPITCH]     (GEMM1 A: b x k)
    bf16* sB = sStage + BT * SAPITCH;  // [INTERX][SAPITCH] (GEMM1 B^T: i x k)
    bf16* sW = sStage;                 // [64][HPITCH]      (GEMM2 B^T chunk: d x i) 8704 elems

    // ------- fold bias + BN into per-column scale/shift (once per block) -------
    for (int i = tid; i < INTERX; i += NTHREADS) {
        const int o = c * INTERX + i;
        float s = g1[o] * rsqrtf(v1[o] + EPSX);
        sScale1[i] = s;
        sShift1[i] = (b1[o] - m1[o]) * s + be1[o];
    }
    for (int d = tid; d < DIMX; d += NTHREADS) {
        const int o = c * DIMX + d;
        float s = g2[o] * rsqrtf(v2[o] + EPSX);
        sScale2[d] = s;
        sShift2[d] = (b2[o] - m2[o]) * s + be2[o];
    }
    __syncthreads();

    // =======================================================================
    // GEMM1: H(128x128) = Xc(128x512) * W1_c^T , k-chunked by 32
    // =======================================================================
    v8f acc[8];
    #pragma unroll
    for (int n = 0; n < 8; ++n) acc[n] = v8f{0.f,0.f,0.f,0.f,0.f,0.f,0.f,0.f};

    for (int k0 = 0; k0 < DIMX; k0 += KC) {
        // --- stage A: x[(b0+r), k0+kk, c] -> sA[r][kk]  (4B gathers, stride C)
        {
            const int r = tid & 127;
            const int h = (tid >> 7) << 4;                 // 0 or 16
            const float* gx = x + ((size_t)(b0 + r) * DIMX + (k0 + h)) * CBR + c;
            bf16* dst = sA + r * SAPITCH + h;
            #pragma unroll
            for (int kk = 0; kk < 16; ++kk)
                dst[kk] = (bf16)gx[(size_t)kk * CBR];
        }
        // --- stage B^T: W1[c][i][k0+kk] -> sB[i][kk]   (coalesced float4)
        {
            const int i  = tid >> 1;
            const int kq = (tid & 1) << 4;                 // 0 or 16
            const float4* gw = (const float4*)(W1 + ((size_t)c * INTERX + i) * DIMX + k0 + kq);
            bf16* dst = sB + i * SAPITCH + kq;
            #pragma unroll
            for (int q = 0; q < 4; ++q) {
                float4 w = gw[q];
                dst[q*4+0] = (bf16)w.x; dst[q*4+1] = (bf16)w.y;
                dst[q*4+2] = (bf16)w.z; dst[q*4+3] = (bf16)w.w;
            }
        }
        __syncthreads();

        v16bf af = frag_a(sA + (wave * 16) * SAPITCH, SAPITCH, lane);
        #pragma unroll
        for (int n = 0; n < 8; ++n) {
            v16bf bfr = frag_b(sB + (n * 16) * SAPITCH, SAPITCH, lane);
            acc[n] = __builtin_amdgcn_wmma_f32_16x16x32_bf16(
                         false, af, false, bfr, (short)0, acc[n], false, false);
        }
        __syncthreads();
    }

    // --------- BN1 + ReLU, bf16, into sH (each wave writes its own strip) -----
    {
        const int mlo  = (lane >> 4) << 3;   // +8 row offset for upper half-lanes
        const int ncol = lane & 15;
        #pragma unroll
        for (int n = 0; n < 8; ++n) {
            const int col = n * 16 + ncol;
            const float s  = sScale1[col];
            const float sh = sShift1[col];
            #pragma unroll
            for (int r = 0; r < 8; ++r) {
                float h = fmaf(acc[n][r], s, sh);
                h = fmaxf(h, 0.f);
                sH[wave * 16 + mlo + r][col] = (bf16)h;
            }
        }
    }

    // A-fragments of H for GEMM2 (each wave reads only rows it just wrote;
    // same-wave DS ops are in-order, compiler inserts s_wait_dscnt).
    v16bf hfrag[4];
    #pragma unroll
    for (int kf = 0; kf < 4; ++kf)
        hfrag[kf] = frag_a(&sH[wave * 16][kf * 32], HPITCH, lane);

    // =======================================================================
    // GEMM2: Y(128x512) = H(128x128) * W2_c^T , 8 output-column chunks of 64
    // =======================================================================
    for (int nc = 0; nc < 8; ++nc) {
        const int d0 = nc * 64;
        __syncthreads();   // previous chunk's reads of sW done before restage
        // --- stage B^T chunk: W2[c][d0+r][i] -> sW[r][i]  (coalesced float4)
        {
            const int r  = tid >> 2;              // 0..63
            const int iq = (tid & 3) << 5;        // 0,32,64,96
            const float4* gw = (const float4*)(W2 + ((size_t)c * DIMX + d0 + r) * INTERX + iq);
            bf16* dst = sW + r * HPITCH + iq;
            #pragma unroll
            for (int q = 0; q < 8; ++q) {
                float4 w = gw[q];
                dst[q*4+0] = (bf16)w.x; dst[q*4+1] = (bf16)w.y;
                dst[q*4+2] = (bf16)w.z; dst[q*4+3] = (bf16)w.w;
            }
        }
        __syncthreads();

        v8f acc2[4];
        #pragma unroll
        for (int nt = 0; nt < 4; ++nt) acc2[nt] = v8f{0.f,0.f,0.f,0.f,0.f,0.f,0.f,0.f};

        #pragma unroll
        for (int nt = 0; nt < 4; ++nt) {
            #pragma unroll
            for (int kf = 0; kf < 4; ++kf) {
                v16bf bfr = frag_b(sW + (nt * 16) * HPITCH + kf * 32, HPITCH, lane);
                acc2[nt] = __builtin_amdgcn_wmma_f32_16x16x32_bf16(
                               false, hfrag[kf], false, bfr, (short)0, acc2[nt], false, false);
            }
        }

        // ------- BN2 + sigmoid + transposed scatter to out[b, d, c] ----------
        const int mlo  = (lane >> 4) << 3;
        const int ncol = lane & 15;
        #pragma unroll
        for (int nt = 0; nt < 4; ++nt) {
            const int d  = d0 + nt * 16 + ncol;
            const float s  = sScale2[d];
            const float sh = sShift2[d];
            #pragma unroll
            for (int r = 0; r < 8; ++r) {
                float yv = fmaf(acc2[nt][r], s, sh);
                float w  = 1.f / (1.f + __expf(-yv));
                const int b = b0 + wave * 16 + mlo + r;
                out[((size_t)b * DIMX + d) * CBR + c] = w;
            }
        }
    }
}

extern "C" void kernel_launch(void* const* d_in, const int* in_sizes, int n_in,
                              void* d_out, int out_size, void* d_ws, size_t ws_size,
                              hipStream_t stream) {
    const float* x   = (const float*)d_in[0];
    const float* W1  = (const float*)d_in[1];
    const float* b1  = (const float*)d_in[2];
    const float* g1  = (const float*)d_in[3];
    const float* be1 = (const float*)d_in[4];
    const float* m1  = (const float*)d_in[5];
    const float* v1  = (const float*)d_in[6];
    const float* W2  = (const float*)d_in[7];
    const float* b2  = (const float*)d_in[8];
    const float* g2  = (const float*)d_in[9];
    const float* be2 = (const float*)d_in[10];
    const float* m2  = (const float*)d_in[11];
    const float* v2  = (const float*)d_in[12];
    float* out = (float*)d_out;

    const int Bsz = in_sizes[0] / (DIMX * CBR);   // 2048
    dim3 grid(CBR, Bsz / BT);                     // (64, 16); c fastest for L2 reuse
    AFM_54022098649750_kernel<<<grid, NTHREADS, 0, stream>>>(
        x, W1, b1, g1, be1, m1, v1, W2, b2, g2, be2, m2, v2, out);
}
